// QuantumCharacterMatrix_55886114455825
// MI455X (gfx1250) — compile-verified, matrix operand order checked by reading, not run on hardware
//
#include <hip/hip_runtime.h>

#define DHID 512
#define PI_F 3.14159265358979f

typedef __attribute__((ext_vector_type(16))) _Float16 v16h;
typedef __attribute__((ext_vector_type(8)))  _Float16 v8h;
typedef __attribute__((ext_vector_type(8)))  float    v8f;

static __device__ __forceinline__ v16h load_frag(const _Float16* p) {
    v8h lo = *(const v8h*)p;          // global_load_b128
    v8h hi = *(const v8h*)(p + 16);   // global_load_b128
    return __builtin_shufflevector(lo, hi,
        0,1,2,3,4,5,6,7,8,9,10,11,12,13,14,15);
}

static __device__ __forceinline__ v8f wmma_f16(v16h a, v16h b, v8f c) {
    return __builtin_amdgcn_wmma_f32_16x16x32_f16(false, a, false, b,
                                                  (short)0, c, false, false);
}

// ---------------------------------------------------------------------------
// Kernel 0: convert W (f32, [512,512] row-major) to f16 in workspace.
// ---------------------------------------------------------------------------
__global__ void convW(const float* __restrict__ W, _Float16* __restrict__ Wh) {
    int i = blockIdx.x * 256 + threadIdx.x;
    Wh[i] = (_Float16)W[i];
}

// ---------------------------------------------------------------------------
// Kernel 1: build A = rotated/normalized real state, f16 [N,512].
// One wave32 per token; 8 tokens per 256-thread block.
// d-only transcendentals (w1,w2,filt) precomputed once per block into LDS.
// ---------------------------------------------------------------------------
__global__ void __launch_bounds__(256)
build_A(const int* __restrict__ char_idx, const int* __restrict__ positions,
        _Float16* __restrict__ Ah, int N) {
    __shared__ float sw1[DHID], sw2[DHID], sfr[DHID], sfi[DHID];
    for (int d = threadIdx.x; d < DHID; d += 256) {
        float df = (float)d;
        sw1[d] = 0.6f + 0.4f * __sinf(0.1f * df);
        sw2[d] = 0.4f + 0.6f * __cosf(0.15f * df);
        float kv = df + 1.0f;
        float pa = 1.5f * atanf(logf(kv + 1e-10f));   // ALPHA * atan(log(k))
        float pb = 0.8f * __sinf(0.1f * kv);          // BETA  * sin(0.1 k)
        sfr[d] = 0.7f * __cosf(pa) + 0.3f * __cosf(pb);
        sfi[d] = 0.7f * __sinf(pa) + 0.3f * __sinf(pb);
    }
    __syncthreads();

    const int wave = threadIdx.x >> 5;
    const int lane = threadIdx.x & 31;
    const int n = blockIdx.x * 8 + wave;
    if (n >= N) return;

    const float lam = (float)char_idx[n] / 96.0f;
    const float t   = (float)positions[n] / 100.0f;
    const float omega = 2.0f * PI_F * 1.5f;           // 2*pi*ALPHA
    const float kwav  = 2.0f * PI_F / 1.7f;           // 2*pi/FRACTAL

    float ph1 = omega * t - kwav * lam + 0.8f * lam * lam;
    float ph2 = omega * t * 1.5f - kwav * lam * 0.7f + 0.8f * lam * lam * 1.3f;
    float a1 = __sinf(omega * t + 1.5f * lam);
    float a2 = __cosf(omega * t * 0.8f + 1.5f * lam * 1.2f);
    float w1r = a1 * __cosf(ph1), w1i = a1 * __sinf(ph1);
    float w2r = a2 * __cosf(ph2), w2i = a2 * __sinf(ph2);
    const float c1 = (2.0f * PI_F / 512.0f) * lam;
    const float c2 = (4.0f * PI_F / 512.0f) * 1.7f * lam;

    float sre[16], sim[16];
    float nrm = 0.0f;
#pragma unroll
    for (int j = 0; j < 16; ++j) {
        int d = lane + 32 * j;
        float df = (float)d;
        float m1s = __sinf(c1 * df), m1c = __cosf(c1 * df);
        float m2s = __sinf(c2 * df), m2c = __cosf(c2 * df);
        float re = sw1[d] * (w1r * m1c - w1i * m1s) + sw2[d] * (w2r * m2c - w2i * m2s);
        float im = sw1[d] * (w1r * m1s + w1i * m1c) + sw2[d] * (w2r * m2s + w2i * m2c);
        float fr = sfr[d], fi = sfi[d];
        float rr = re * fr - im * fi;
        float ii = re * fi + im * fr;
        sre[j] = rr; sim[j] = ii;
        nrm += rr * rr + ii * ii;
    }
#pragma unroll
    for (int off = 16; off > 0; off >>= 1)
        nrm += __shfl_xor(nrm, off, 32);
    const float inv = 1.0f / (sqrtf(nrm) + 1e-8f);
    const float ct = 0.99500416527803f, st = 0.09983341664683f;  // cos/sin(0.1)
#pragma unroll
    for (int j = 0; j < 16; ++j) {
        float x = (sre[j] * ct - sim[j] * st) * inv;
        Ah[(size_t)n * DHID + lane + 32 * j] = (_Float16)x;
    }
}

// ---------------------------------------------------------------------------
// Kernel 2: fused GEMM (x @ W^T + b) via v_wmma_f32_16x16x32_f16
//           + LayerNorm + noise + quaternion expansion.
// One block = 16 token rows; 8 waves, each wave = 16x64 C tile.
// K loop is software-pipelined (1-stage double buffer) so next-iteration
// fragment loads overlap the current WMMA cluster.
// ---------------------------------------------------------------------------
__global__ void __launch_bounds__(256)
gemm_ln_quat(const _Float16* __restrict__ Ah, const _Float16* __restrict__ Wh,
             const float* __restrict__ bias, const float* __restrict__ gamma,
             const float* __restrict__ betaln, const float* __restrict__ noise,
             const int* __restrict__ char_idx, const float* __restrict__ sem,
             float* __restrict__ out) {
    __shared__ float xt[16][DHID];              // 32 KB row tile
    __shared__ float redS[16][16], redQ[16][16];
    __shared__ float muS[16], rsS[16];
    __shared__ float rowCpm[16], rowSw0[16], rowSw1[16], rowSw2[16];

    const int tid  = threadIdx.x;
    const int wave = tid >> 5;
    const int lane = tid & 31;
    const int n_base   = blockIdx.x * 16;
    const int col_base = wave * 64;
    const int r     = lane & 15;           // row (A) / col (B) within 16
    const int khalf = (lane >> 4) * 8;     // K sub-half select per ISA layout

    v8f zero = {};
    v8f acc0 = zero, acc1 = zero, acc2 = zero, acc3 = zero;

    const _Float16* arow = Ah + (size_t)(n_base + r) * DHID + khalf;
    const _Float16* b0 = Wh + (size_t)(col_base + 0  + r) * DHID + khalf;
    const _Float16* b1 = Wh + (size_t)(col_base + 16 + r) * DHID + khalf;
    const _Float16* b2 = Wh + (size_t)(col_base + 32 + r) * DHID + khalf;
    const _Float16* b3 = Wh + (size_t)(col_base + 48 + r) * DHID + khalf;

    // Prologue: stage K-chunk 0.
    v16h aC  = load_frag(arow);
    v16h bC0 = load_frag(b0);
    v16h bC1 = load_frag(b1);
    v16h bC2 = load_frag(b2);
    v16h bC3 = load_frag(b3);

#pragma unroll 3
    for (int k0 = 0; k0 < DHID - 32; k0 += 32) {
        const int kn = k0 + 32;
        // Issue next-stage loads before consuming current fragments.
        v16h aN  = load_frag(arow + kn);
        v16h bN0 = load_frag(b0 + kn);
        v16h bN1 = load_frag(b1 + kn);
        v16h bN2 = load_frag(b2 + kn);
        v16h bN3 = load_frag(b3 + kn);

        acc0 = wmma_f16(aC, bC0, acc0);
        acc1 = wmma_f16(aC, bC1, acc1);
        acc2 = wmma_f16(aC, bC2, acc2);
        acc3 = wmma_f16(aC, bC3, acc3);

        aC = aN; bC0 = bN0; bC1 = bN1; bC2 = bN2; bC3 = bN3;
    }
    // Epilogue K-chunk.
    acc0 = wmma_f16(aC, bC0, acc0);
    acc1 = wmma_f16(aC, bC1, acc1);
    acc2 = wmma_f16(aC, bC2, acc2);
    acc3 = wmma_f16(aC, bC3, acc3);

    // Scatter C (+bias) to LDS per 16x16 f32 C layout:
    // VGPR e: lanes 0-15 -> M=e, N=lane; lanes 16-31 -> M=e+8, N=lane-16.
    const int rowhi = (lane >> 4) * 8;
    {
        v8f accs[4] = {acc0, acc1, acc2, acc3};
#pragma unroll
        for (int t = 0; t < 4; ++t) {
            int col = col_base + 16 * t + r;
            float bv = bias[col];
#pragma unroll
            for (int e = 0; e < 8; ++e)
                xt[rowhi + e][col] = accs[t][e] + bv;
        }
    }
    __syncthreads();

    // LayerNorm partial sums: 16 threads per row, 32 cols per thread.
    {
        int row = tid & 15, chunk = tid >> 4;
        float s = 0.0f, q = 0.0f;
        int c0 = chunk * 32;
#pragma unroll
        for (int c = 0; c < 32; ++c) {
            float v = xt[row][c0 + c];
            s += v; q += v * v;
        }
        redS[row][chunk] = s; redQ[row][chunk] = q;
    }
    __syncthreads();
    if (tid < 16) {
        float s = 0.0f, q = 0.0f;
#pragma unroll
        for (int c = 0; c < 16; ++c) { s += redS[tid][c]; q += redQ[tid][c]; }
        float mu = s * (1.0f / 512.0f);
        float var = q * (1.0f / 512.0f) - mu * mu;
        muS[tid] = mu;
        rsS[tid] = rsqrtf(var + 1e-5f);
        // Per-row quaternion constants (hoisted out of the epilogue loop).
        int ci = char_idx[n_base + tid];
        rowCpm[tid] = (float)ci * 0.01f;
        const float* swp = sem + 4 * ci;
        rowSw0[tid] = swp[0];
        rowSw1[tid] = swp[1];
        rowSw2[tid] = swp[2];
    }
    __syncthreads();

    // Epilogue: y + quaternion expansion, coalesced float4 stores.
    const float TWO_PI_D = 2.0f * PI_F / 512.0f;
    for (int idx = tid; idx < 16 * DHID; idx += 256) {
        int row = idx >> 9, m = idx & 511;
        int n = n_base + row;
        float y = (xt[row][m] - muS[row]) * rsS[row] * gamma[m] + betaln[m]
                + 0.01f * noise[(size_t)n * DHID + m];
        float cpm = rowCpm[row];
        float phase = TWO_PI_D * (float)m;
        float4 o;
        o.x = y;
        o.y = y * rowSw0[row] * __cosf(phase + cpm);
        o.z = y * rowSw1[row] * __sinf(phase + cpm * 1.3f);
        o.w = y * rowSw2[row] * __cosf(2.0f * phase + cpm * 0.7f);
        *(float4*)(out + ((size_t)n * DHID + m) * 4) = o;
    }
}

// ---------------------------------------------------------------------------
extern "C" void kernel_launch(void* const* d_in, const int* in_sizes, int n_in,
                              void* d_out, int out_size, void* d_ws, size_t ws_size,
                              hipStream_t stream) {
    const int*   char_idx  = (const int*)d_in[0];
    const int*   positions = (const int*)d_in[1];
    const float* W         = (const float*)d_in[2];
    const float* b         = (const float*)d_in[3];
    const float* gamma     = (const float*)d_in[4];
    const float* betaln    = (const float*)d_in[5];
    const float* noise     = (const float*)d_in[6];
    const float* sem       = (const float*)d_in[7];
    float*       out       = (float*)d_out;
    const int N = in_sizes[0];   // 16384 tokens

    _Float16* Ah = (_Float16*)d_ws;                      // N*512 halves = 16 MB
    _Float16* Wh = Ah + (size_t)N * DHID;                // 512*512 halves = 512 KB

    convW<<<(DHID * DHID) / 256, 256, 0, stream>>>(W, Wh);
    build_A<<<(N + 7) / 8, 256, 0, stream>>>(char_idx, positions, Ah, N);
    gemm_ln_quat<<<N / 16, 256, 0, stream>>>(Ah, Wh, b, gamma, betaln, noise,
                                             char_idx, sem, out);
}
